// TrafficNetworkDQN_85401129713921
// MI455X (gfx1250) — matrix operand order repaired and checked by reading, not compiled, not run on hardware
//
#include <hip/hip_runtime.h>
#include <hip/hip_bf16.h>

typedef __attribute__((ext_vector_type(16))) _Float16 v16h;
typedef __attribute__((ext_vector_type(8)))  float    v8f;

// ---------------------------------------------------------------------------
// WMMA 16x16x32 f16 fragment layout helpers (CDNA5 ISA 7.12.2, wave32)
// A (16xK, f16), this file's K order, not the ISA table's: lanes 0-15 hold M=0..15 with K 0..15 (elem i -> K = i);
//                lanes 16-31 hold the same rows with K 16..31
//                (elem i -> K = 16 + i).
// B (Kx16, f16), same K order as A: lane = column N (mod 16); lanes 0-15 hold K=0..15,
//                lanes 16-31 hold K=16..31, element i -> K = k0 + i.
// C/D (16x16, f32): lane = column N (mod 16); VGPR r -> row M = r (lanes
//                0-15) or M = r + 8 (lanes 16-31).
// ---------------------------------------------------------------------------

__device__ __forceinline__ int a_kidx(int lane, int i) {
  const int hi = (lane >> 4) & 1;
  return (hi ? 16 : 0) + i;
}

// Build a resident B fragment from a row-major f32 weight W[Klim x ld],
// taking columns [colbase, colbase+16). Zero-pads K -> 32 and N -> Nlim.
__device__ __forceinline__ v16h load_bfrag_global(const float* __restrict__ W,
                                                  int ld, int colbase,
                                                  int Klim, int Nlim, int lane) {
  const int n  = colbase + (lane & 15);
  const int k0 = (lane < 16) ? 0 : 16;
  v16h b;
#pragma unroll
  for (int i = 0; i < 16; ++i) {
    const int k = k0 + i;
    const float v = (k < Klim && n < Nlim) ? W[k * ld + n] : 0.0f;
    b[i] = (_Float16)v;
  }
  return b;
}

__device__ __forceinline__ v8f wmma_f16(v16h a, v16h b, v8f c) {
  // (neg_a, A, neg_b, B, c_mod, C, reuse_a, reuse_b)
  return __builtin_amdgcn_wmma_f32_16x16x32_f16(false, a, false, b,
                                                (short)0, c, false, false);
}

__device__ __forceinline__ void bias_relu(v8f& c, float bn) {
#pragma unroll
  for (int r = 0; r < 8; ++r) c[r] = fmaxf(c[r] + bn, 0.0f);
}

// Scatter a C fragment (16 rows x 16 cols) into a per-wave LDS tile.
__device__ __forceinline__ void store_c_lds(float* lds, int stride, int coloff,
                                            const v8f& c, int lane) {
  const int nn    = (lane & 15) + coloff;
  const int mbase = (lane < 16) ? 0 : 8;
#pragma unroll
  for (int r = 0; r < 8; ++r) lds[(mbase + r) * stride + nn] = c[r];
}

// Gather an A fragment (16 rows x Klim, zero-padded to K=32) from LDS.
__device__ __forceinline__ v16h load_afrag_lds(const float* lds, int stride,
                                               int Klim, int lane) {
  const int m = lane & 15;
  v16h a;
#pragma unroll
  for (int i = 0; i < 16; ++i) {
    const int k = a_kidx(lane, i);
    a[i] = (k < Klim) ? (_Float16)lds[m * stride + k] : (_Float16)0.0f;
  }
  return a;
}

#define TILE_LDS_F32 (16 * 32)  // 2 KB per 16-row sub-tile

__global__ __launch_bounds__(128) void traffic_dqn_kernel(
    const float* __restrict__ lf, const int* __restrict__ sig,
    const float* __restrict__ le_w1, const float* __restrict__ le_b1,
    const float* __restrict__ le_w2, const float* __restrict__ le_b2,
    const float* __restrict__ emb,
    const float* __restrict__ ls_w1, const float* __restrict__ ls_b1,
    const float* __restrict__ ls_w2, const float* __restrict__ ls_b2,
    const float* __restrict__ ls_w3, const float* __restrict__ ls_b3,
    const float* __restrict__ as_w1, const float* __restrict__ as_b1,
    const float* __restrict__ as_w2, const float* __restrict__ as_b2,
    const float* __restrict__ as_w3, const float* __restrict__ as_b3,
    float* __restrict__ out_scores, float* __restrict__ out_av, int ntiles) {
  __shared__ float smem[4 * 2 * TILE_LDS_F32];  // 4 waves x 2 sub-tiles
  const int lane = threadIdx.x & 31;
  const int wave = threadIdx.x >> 5;
  float* lds0 = &smem[wave * 2 * TILE_LDS_F32];
  float* ldsu[2] = {lds0, lds0 + TILE_LDS_F32};

  // ---- loop-invariant weight fragments, resident in VGPRs ----------------
  const v16h B_le1  = load_bfrag_global(le_w1, 16, 0, 5, 16, lane);
  const v16h B_le2  = load_bfrag_global(le_w2, 8, 0, 16, 8, lane);
  const v16h B_ls1a = load_bfrag_global(ls_w1, 32, 0, 16, 32, lane);
  const v16h B_ls1b = load_bfrag_global(ls_w1, 32, 16, 16, 32, lane);
  const v16h B_ls2  = load_bfrag_global(ls_w2, 16, 0, 32, 16, lane);
  const v16h B_as1a = load_bfrag_global(as_w1, 32, 0, 16, 32, lane);
  const v16h B_as1b = load_bfrag_global(as_w1, 32, 16, 16, 32, lane);
  const v16h B_as2  = load_bfrag_global(as_w2, 16, 0, 32, 16, lane);

  const int n = lane & 15;
  const float b_le1  = le_b1[n];
  const float b_le2  = (n < 8) ? le_b2[n] : 0.0f;
  const float b_ls1a = ls_b1[n];
  const float b_ls1b = ls_b1[16 + n];
  const float b_ls2  = ls_b2[n];
  const float b_as1a = as_b1[n];
  const float b_as1b = as_b1[16 + n];
  const float b_as2  = as_b2[n];

  // Uniform-address scalars -> SGPR resident.
  float w3l[16], w3a0[16], w3a1[16];
#pragma unroll
  for (int k = 0; k < 16; ++k) {
    w3l[k]  = ls_w3[k];
    w3a0[k] = as_w3[k * 2 + 0];
    w3a1[k] = as_w3[k * 2 + 1];
  }
  const float b3l  = ls_b3[0];
  const float b3a0 = as_b3[0];
  const float b3a1 = as_b3[1];

  const int waveGlobal = blockIdx.x * (blockDim.x >> 5) + wave;
  const int nwaves     = gridDim.x * (blockDim.x >> 5);

  for (int t = waveGlobal; t < ntiles; t += nwaves) {
    const int rb[2] = {t * 32, t * 32 + 16};  // two 16-row sub-tiles

    // ---- link encoder layer 1: relu(X[16x5] @ le_w1[5x16] + b1) ---------
#pragma unroll
    for (int u = 0; u < 2; ++u) {
      v16h a0;
      const int m = lane & 15;
      const float* rowp = lf + (size_t)(rb[u] + m) * 5;
#pragma unroll
      for (int i = 0; i < 16; ++i) {
        const int k = a_kidx(lane, i);
        a0[i] = (k < 5) ? (_Float16)rowp[k] : (_Float16)0.0f;
      }
      v8f c = {};
      c = wmma_f16(a0, B_le1, c);
      bias_relu(c, b_le1);
      store_c_lds(ldsu[u], 32, 0, c, lane);
    }
    asm volatile("s_wait_dscnt 0" ::: "memory");

    // ---- link encoder layer 2 + comb assembly ----------------------------
#pragma unroll
    for (int u = 0; u < 2; ++u) {
      const v16h a1 = load_afrag_lds(ldsu[u], 32, 16, lane);
      v8f c2 = {};
      c2 = wmma_f16(a1, B_le2, c2);
      bias_relu(c2, b_le2);  // padded cols get bias 0, relu keeps 0
      const int mbase = (lane < 16) ? 0 : 8;
      if (n < 8) {  // cols 0..7 = link_emb
#pragma unroll
        for (int r = 0; r < 8; ++r) ldsu[u][(mbase + r) * 32 + n] = c2[r];
      } else {      // cols 8..15 = emb[signal_state]
#pragma unroll
        for (int r = 0; r < 8; ++r) {
          const int s = sig[rb[u] + mbase + r];
          ldsu[u][(mbase + r) * 32 + n] = emb[s * 8 + (n - 8)];
        }
      }
    }
    asm volatile("s_wait_dscnt 0" ::: "memory");

    v16h a2[2];  // comb fragments, reused by both heads
#pragma unroll
    for (int u = 0; u < 2; ++u) a2[u] = load_afrag_lds(ldsu[u], 32, 16, lane);

    // ---- link-selector head: 16 -> 32 -> 16 -> 1 -------------------------
#pragma unroll
    for (int u = 0; u < 2; ++u) {
      v8f h1a = {}, h1b = {};
      h1a = wmma_f16(a2[u], B_ls1a, h1a);
      h1b = wmma_f16(a2[u], B_ls1b, h1b);
      bias_relu(h1a, b_ls1a);
      bias_relu(h1b, b_ls1b);
      store_c_lds(ldsu[u], 32, 0, h1a, lane);
      store_c_lds(ldsu[u], 32, 16, h1b, lane);
    }
    asm volatile("s_wait_dscnt 0" ::: "memory");
#pragma unroll
    for (int u = 0; u < 2; ++u) {
      const v16h a3 = load_afrag_lds(ldsu[u], 32, 32, lane);  // full K=32
      v8f h2 = {};
      h2 = wmma_f16(a3, B_ls2, h2);
      bias_relu(h2, b_ls2);
      store_c_lds(ldsu[u], 32, 0, h2, lane);
    }
    asm volatile("s_wait_dscnt 0" ::: "memory");
    if (lane < 16) {
#pragma unroll
      for (int u = 0; u < 2; ++u) {
        float acc = b3l;
#pragma unroll
        for (int k = 0; k < 16; ++k) acc += ldsu[u][lane * 32 + k] * w3l[k];
        out_scores[rb[u] + lane] = acc;
      }
    }
    asm volatile("s_wait_dscnt 0" ::: "memory");  // reads done before reuse

    // ---- action head: 16 -> 32 -> 16 -> 2 (reuses a2) --------------------
#pragma unroll
    for (int u = 0; u < 2; ++u) {
      v8f g1a = {}, g1b = {};
      g1a = wmma_f16(a2[u], B_as1a, g1a);
      g1b = wmma_f16(a2[u], B_as1b, g1b);
      bias_relu(g1a, b_as1a);
      bias_relu(g1b, b_as1b);
      store_c_lds(ldsu[u], 32, 0, g1a, lane);
      store_c_lds(ldsu[u], 32, 16, g1b, lane);
    }
    asm volatile("s_wait_dscnt 0" ::: "memory");
#pragma unroll
    for (int u = 0; u < 2; ++u) {
      const v16h a4 = load_afrag_lds(ldsu[u], 32, 32, lane);
      v8f g2 = {};
      g2 = wmma_f16(a4, B_as2, g2);
      bias_relu(g2, b_as2);
      store_c_lds(ldsu[u], 32, 0, g2, lane);
    }
    asm volatile("s_wait_dscnt 0" ::: "memory");
    {
      const int m = lane & 15;
      const int j = (lane < 16) ? 0 : 1;
#pragma unroll
      for (int u = 0; u < 2; ++u) {
        float acc = (j == 0) ? b3a0 : b3a1;
#pragma unroll
        for (int k = 0; k < 16; ++k)
          acc += ldsu[u][m * 32 + k] * ((j == 0) ? w3a0[k] : w3a1[k]);
        out_av[(rb[u] + m) * 2 + j] = acc;
      }
    }
    asm volatile("s_wait_dscnt 0" ::: "memory");  // reads done before next tile
  }
}

extern "C" void kernel_launch(void* const* d_in, const int* in_sizes, int n_in,
                              void* d_out, int out_size, void* d_ws, size_t ws_size,
                              hipStream_t stream) {
  (void)in_sizes; (void)n_in; (void)out_size; (void)d_ws; (void)ws_size;
  const float* lf    = (const float*)d_in[0];
  const int*   sig   = (const int*)d_in[1];
  const float* le_w1 = (const float*)d_in[2];
  const float* le_b1 = (const float*)d_in[3];
  const float* le_w2 = (const float*)d_in[4];
  const float* le_b2 = (const float*)d_in[5];
  const float* emb   = (const float*)d_in[6];
  const float* ls_w1 = (const float*)d_in[7];
  const float* ls_b1 = (const float*)d_in[8];
  const float* ls_w2 = (const float*)d_in[9];
  const float* ls_b2 = (const float*)d_in[10];
  const float* ls_w3 = (const float*)d_in[11];
  const float* ls_b3 = (const float*)d_in[12];
  const float* as_w1 = (const float*)d_in[13];
  const float* as_b1 = (const float*)d_in[14];
  const float* as_w2 = (const float*)d_in[15];
  const float* as_b2 = (const float*)d_in[16];
  const float* as_w3 = (const float*)d_in[17];
  const float* as_b3 = (const float*)d_in[18];

  const int NROWS  = 4096 * 256;
  const int ntiles = NROWS / 32;  // 32768 32-row tiles (2 x 16-row sub-tiles)
  float* out_scores = (float*)d_out;
  float* out_av     = out_scores + NROWS;

  dim3 block(128);  // 4 waves per block
  dim3 grid(1024);  // 4096 waves -> 8 tiles per wave
  hipLaunchKernelGGL(traffic_dqn_kernel, grid, block, 0, stream,
                     lf, sig, le_w1, le_b1, le_w2, le_b2, emb,
                     ls_w1, ls_b1, ls_w2, ls_b2, ls_w3, ls_b3,
                     as_w1, as_b1, as_w2, as_b2, as_w3, as_b3,
                     out_scores, out_av, ntiles);
}
